// AttentionNet_80650895885001
// MI455X (gfx1250) — matrix operand order repaired and checked by reading, not compiled, not run on hardware
//
#include <hip/hip_runtime.h>
#include <math.h>

typedef __attribute__((ext_vector_type(8)))  _Float16     v8h_t;
typedef __attribute__((ext_vector_type(16))) _Float16     v16h_t;
typedef __attribute__((ext_vector_type(8)))  float        v8f_t;
typedef __attribute__((ext_vector_type(4)))  unsigned int u32x4;
typedef __attribute__((ext_vector_type(8)))  int          i32x8;
typedef __attribute__((ext_vector_type(4)))  int          i32x4;

#define D_   128
#define H_   4
#define DK_  32
#define DFF_ 512
#define G_   2048
#define B_   512
#define TG   32      // rows of memory per tile
#define NT   (G_ / TG)
#define NKV  256     // K(128) | V(128) output columns

// ---- shared-memory arena (manually aliased, 63008 B) ----
#define SA0_OFF   0        // phase1: f16 tile buf 0  (8 KB)
#define SA1_OFF   8192     // phase1: f16 tile buf 1  (8 KB)
#define SC_OFF    16384    // phase1: K|V f32 tile    (32 KB)
#define SA3_OFF   0        // phase3: f16 tile        (8 KB)
#define STG_OFF   8192     // phase3: f32 TDM staging (16 KB)
#define SKP_OFF   24576    // phase3: Kp f32 tile     (16 KB)
#define SUP_OFF   49152    // pointer logits          (8 KB)
#define SVEC_OFF  57344
#define SQ_OFF    57856
#define SHEADS_OFF 58368
#define SHRES_OFF 58880
#define SHN_OFF   59392
#define SDEC_OFF  59904
#define SQP_OFF   60416
#define SHID_OFF  60928   // 2 KB
#define SRED_OFF  62976   // 32 B
#define SMEM_BYTES 63008

#if defined(__has_builtin)
#if __has_builtin(__builtin_amdgcn_tensor_load_to_lds) && __has_builtin(__builtin_amdgcn_s_wait_tensorcnt)
#define HAS_TDM 1
#endif
#endif
#ifndef HAS_TDM
#define HAS_TDM 0
#endif

// ---------------- reductions ----------------
__device__ __forceinline__ float waveReduceSum(float v) {
#pragma unroll
  for (int m = 16; m >= 1; m >>= 1) v += __shfl_xor(v, m, 32);
  return v;
}
__device__ __forceinline__ float waveReduceMax(float v) {
#pragma unroll
  for (int m = 16; m >= 1; m >>= 1) v = fmaxf(v, __shfl_xor(v, m, 32));
  return v;
}
__device__ __forceinline__ float blockReduceSum(float v, float* sred) {
  int tid = threadIdx.x, wave = tid >> 5, lane = tid & 31;
  v = waveReduceSum(v);
  if (lane == 0) sred[wave] = v;
  __syncthreads();
  float t = (lane < 8) ? sred[lane] : 0.f;
  t = waveReduceSum(t);
  __syncthreads();
  return t;
}
__device__ __forceinline__ float blockReduceMax(float v, float* sred) {
  int tid = threadIdx.x, wave = tid >> 5, lane = tid & 31;
  v = waveReduceMax(v);
  if (lane == 0) sred[wave] = v;
  __syncthreads();
  float t = (lane < 8) ? sred[lane] : -3.0e38f;
  t = waveReduceMax(t);
  __syncthreads();
  return t;
}

#if HAS_TDM
// Issue one TDM load of a 32x128 f32 tile (row-major, stride 128) into LDS.
// D# per CDNA5 ISA ch.8: group0 = {count/type/addr}, group1 = {dims/tile/strides}.
__device__ __forceinline__ void tdm_load_tile(const float* gsrc, unsigned int lds_off) {
  unsigned long long ga = (unsigned long long)(uintptr_t)gsrc;
  u32x4 g0;
  g0.x = 1u;                                                  // count=1, user descriptor
  g0.y = lds_off;                                             // lds_addr (bytes)
  g0.z = (unsigned int)ga;                                    // global_addr[31:0]
  g0.w = (unsigned int)((ga >> 32) & 0x1FFFFFFu) | (2u << 30); // addr[56:32] | type=2
  i32x8 g1;
  g1[0] = 0x20000;                 // workgroup_mask=0, data_size=2 (4 bytes)
  g1[1] = (int)(128u << 16);       // tensor_dim0 = 128 (bits 79:48, low half)
  g1[2] = (int)(TG << 16);         // tensor_dim1 = 32   (bits 111:80, low half)
  g1[3] = (int)(128u << 16);       // tile_dim0 = 128    (bits 127:112)
  g1[4] = TG;                      // tile_dim1 = 32     (bits 143:128)
  g1[5] = 128;                     // tensor_dim0_stride = 128 (bits 207:160, low)
  g1[6] = 0;
  g1[7] = 0;
  i32x4 z4 = {0, 0, 0, 0};
#if __clang_major__ >= 23
  i32x8 z8 = {0, 0, 0, 0, 0, 0, 0, 0};
  __builtin_amdgcn_tensor_load_to_lds(g0, g1, z4, z4, z8, 0);
#else
  __builtin_amdgcn_tensor_load_to_lds(g0, g1, z4, z4, 0);
#endif
}
#endif

// ---------------- kernel 0: weight repack (f32 -> transposed f16 panels) ----------------
__global__ void prep_weights(const float* __restrict__ wk, const float* __restrict__ wv,
                             const float* __restrict__ pwk,
                             _Float16* __restrict__ wkvt, _Float16* __restrict__ pkt) {
  int idx = blockIdx.x * 256 + threadIdx.x;
  if (idx < NKV * D_) {
    int col = idx / D_, d = idx % D_;
    float v;
    if (col < 128) { int h = col >> 5, k = col & 31; v = wk[(h * D_ + d) * DK_ + k]; }
    else           { int c = col - 128; int h = c >> 5, k = c & 31; v = wv[(h * D_ + d) * DK_ + k]; }
    wkvt[idx] = (_Float16)v;
  }
  if (idx < D_ * D_) {
    int col = idx / D_, d = idx % D_;
    pkt[idx] = (_Float16)pwk[d * D_ + col];
  }
}

// 16x16x32 f16 WMMA subtile: A from LDS (row-major f16, stride 128), B panel from global f16.
__device__ __forceinline__ v8f_t wmma_tile_128(const _Float16* sArow0,   // &sA[m0*128]
                                               const _Float16* bPanel,   // &W[n0col*128] base incl lane col
                                               int lane) {
  v8f_t acc = {};
  int rowA = lane & 15;
  int koff = (lane >> 4) * 16;
  int kbB  = (lane >> 4) * 16;
#pragma unroll
  for (int kt = 0; kt < 4; ++kt) {
    int k0 = kt * 32;
    v8h_t alo = *(const v8h_t*)&sArow0[rowA * D_ + k0 + koff];
    v8h_t ahi = *(const v8h_t*)&sArow0[rowA * D_ + k0 + koff + 8];
    v16h_t a = __builtin_shufflevector(alo, ahi, 0,1,2,3,4,5,6,7,8,9,10,11,12,13,14,15);
    v16h_t bf = *(const v16h_t*)&bPanel[k0 + kbB];
    acc = __builtin_amdgcn_wmma_f32_16x16x32_f16(false, a, false, bf, (short)0, acc, false, false);
  }
  return acc;
}

// ---------------- main fused kernel: one block = one batch ----------------
__global__ __launch_bounds__(256)
void attn_pointer_kernel(const float* __restrict__ tgt, const float* __restrict__ mem,
                         const unsigned char* __restrict__ mask,
                         const float* __restrict__ ln1w, const float* __restrict__ ln1b,
                         const float* __restrict__ ln2w, const float* __restrict__ ln2b,
                         const float* __restrict__ wq,  const float* __restrict__ wo,
                         const float* __restrict__ w1,  const float* __restrict__ b1,
                         const float* __restrict__ w2,  const float* __restrict__ b2,
                         const float* __restrict__ pwq,
                         const _Float16* __restrict__ wkvt, const _Float16* __restrict__ pkt,
                         float* __restrict__ out) {
  __shared__ __align__(32) unsigned char smem[SMEM_BYTES];
  _Float16* sA0   = (_Float16*)(smem + SA0_OFF);
  _Float16* sA1   = (_Float16*)(smem + SA1_OFF);
  float*    sC    = (float*)(smem + SC_OFF);     // [TG][NKV]
  _Float16* sA3   = (_Float16*)(smem + SA3_OFF); // [TG][128]
  float*    stg   = (float*)(smem + STG_OFF);    // [TG][128]
  float*    sKp   = (float*)(smem + SKP_OFF);    // [TG][128]
  float*    sUp   = (float*)(smem + SUP_OFF);
  float*    sVec  = (float*)(smem + SVEC_OFF);
  float*    sQ    = (float*)(smem + SQ_OFF);
  float*    sHeads= (float*)(smem + SHEADS_OFF);
  float*    sHres = (float*)(smem + SHRES_OFF);
  float*    sHn   = (float*)(smem + SHN_OFF);
  float*    sDec  = (float*)(smem + SDEC_OFF);
  float*    sQp   = (float*)(smem + SQP_OFF);
  float*    sHid  = (float*)(smem + SHID_OFF);
  float*    sRed  = (float*)(smem + SRED_OFF);

  const int b = blockIdx.x;
  const int tid = threadIdx.x, lane = tid & 31, wave = tid >> 5;
  const int r  = tid >> 3;          // tile row handled by this thread (8 lanes/row)
  const int c0 = (tid & 7) * 16;    // 16 contiguous columns per lane

  // ---------- phase 0: LN(tgt), Q projections ----------
  const float* tgtb = tgt + (size_t)b * D_;
  float x0 = (tid < D_) ? tgtb[tid] : 0.f;
  float s  = blockReduceSum(x0, sRed);
  float sq = blockReduceSum(x0 * x0, sRed);
  float mu = s / D_, var = sq / D_ - mu * mu;
  float rstd = rsqrtf(var + 1e-5f);
  if (tid < D_) sVec[tid] = (x0 - mu) * rstd * ln1w[tid] + ln1b[tid];
  __syncthreads();
  if (tid < D_) {
    int h = tid >> 5, k = tid & 31;
    float acc = 0.f;
    for (int d = 0; d < D_; ++d) acc += sVec[d] * wq[(h * D_ + d) * DK_ + k];
    sQ[tid] = acc;
  }
  __syncthreads();

  // ---------- phase 1: pipelined stream: K,V WMMA + online softmax ----------
  float accV = 0.f;                  // head = wave (waves 0..3), channel k = lane
  float mRun = -3.0e38f, lRun = 0.f;
  const float scaleQK = 0.17677669529663689f;   // 1/sqrt(32)

  // prologue: tile 0 -> sA0
  {
    const float* row = mem + ((size_t)b * G_ + r) * D_;
    float x[16]; float ls = 0.f, lsq = 0.f;
#pragma unroll
    for (int i = 0; i < 16; ++i) { x[i] = row[c0 + i]; ls += x[i]; lsq += x[i] * x[i]; }
#pragma unroll
    for (int m = 4; m >= 1; m >>= 1) { ls += __shfl_xor(ls, m, 32); lsq += __shfl_xor(lsq, m, 32); }
    float mu1 = ls / D_, v1 = lsq / D_ - mu1 * mu1, rs1 = rsqrtf(v1 + 1e-5f);
#pragma unroll
    for (int i = 0; i < 16; ++i)
      sA0[r * D_ + c0 + i] = (_Float16)((x[i] - mu1) * rs1 * ln1w[c0 + i] + ln1b[c0 + i]);
  }
  __syncthreads();

  for (int t = 0; t < NT; ++t) {
    const int g0 = t * TG;
    const _Float16* sAcur = (t & 1) ? sA1 : sA0;
    _Float16*       sAnxt = (t & 1) ? sA0 : sA1;
    const bool hasNext = (t + 1 < NT);

    // (a) issue next tile's global loads early; consumed only after WMMA+scores
    float xn[16];
    if (hasNext) {
      const float* rown = mem + ((size_t)b * G_ + g0 + TG + r) * D_;
#pragma unroll
      for (int i = 0; i < 16; ++i) xn[i] = rown[c0 + i];
      if (t + 2 < NT) __builtin_prefetch(rown + TG * D_, 0, 0);
    }

    // (b) WMMA: C(32x256) = A(32x128) x Wkv(128x256); 32 subtiles, 4 per wave
#pragma unroll
    for (int si = 0; si < 4; ++si) {
      int s4 = wave * 4 + si;
      int m0 = (s4 & 1) * 16;
      int n0 = (s4 >> 1) * 16;
      int ncol = n0 + (lane & 15);
      v8f_t acc = wmma_tile_128(&sAcur[m0 * D_], &wkvt[ncol * D_], lane);
      int mrow = m0 + ((lane >> 4) * 8);
#pragma unroll
      for (int i = 0; i < 8; ++i) sC[(mrow + i) * NKV + ncol] = acc[i];
    }
    __syncthreads();

    // (c) online softmax: one wave per head
    if (wave < H_) {
      int h = wave;
      float qk = sQ[h * 32 + lane];
      float sg[TG];
      float tmax = -3.0e38f;
#pragma unroll
      for (int g = 0; g < TG; ++g) {
        float p = qk * sC[g * NKV + h * 32 + lane];
        p = waveReduceSum(p) * scaleQK;
        bool mk = mask[(size_t)b * G_ + g0 + g] != 0;
        sg[g] = mk ? -3.0e38f : p;
        tmax = fmaxf(tmax, sg[g]);
      }
      float newm = fmaxf(mRun, tmax);
      float sc = __expf(mRun - newm);
      lRun *= sc; accV *= sc;
#pragma unroll
      for (int g = 0; g < TG; ++g) {
        float p = __expf(sg[g] - newm);
        lRun += p;
        accV += p * sC[g * NKV + 128 + h * 32 + lane];
      }
      mRun = newm;
    }

    // (d) layernorm the prefetched tile into the other buffer
    if (hasNext) {
      float ls = 0.f, lsq = 0.f;
#pragma unroll
      for (int i = 0; i < 16; ++i) { ls += xn[i]; lsq += xn[i] * xn[i]; }
#pragma unroll
      for (int m = 4; m >= 1; m >>= 1) { ls += __shfl_xor(ls, m, 32); lsq += __shfl_xor(lsq, m, 32); }
      float mu1 = ls / D_, v1 = lsq / D_ - mu1 * mu1, rs1 = rsqrtf(v1 + 1e-5f);
#pragma unroll
      for (int i = 0; i < 16; ++i)
        sAnxt[r * D_ + c0 + i] = (_Float16)((xn[i] - mu1) * rs1 * ln1w[c0 + i] + ln1b[c0 + i]);
    }
    __syncthreads();
  }

  // ---------- phase 2: output proj, residual, LN2, FFN, pointer Qp ----------
  if (wave < H_) sHeads[wave * 32 + lane] = accV / lRun;
  __syncthreads();
  if (tid < D_) {
    float acc = 0.f;
    for (int i = 0; i < D_; ++i) acc += sHeads[i] * wo[i * D_ + tid];
    sHres[tid] = acc + x0;
  }
  __syncthreads();
  float hc = (tid < D_) ? sHres[tid] : 0.f;
  float s2  = blockReduceSum(hc, sRed);
  float sq2 = blockReduceSum(hc * hc, sRed);
  float mu2 = s2 / D_, var2 = sq2 / D_ - mu2 * mu2, rstd2 = rsqrtf(var2 + 1e-5f);
  if (tid < D_) sHn[tid] = (hc - mu2) * rstd2 * ln2w[tid] + ln2b[tid];
  __syncthreads();
  for (int j = tid; j < DFF_; j += 256) {
    float acc = b1[j];
    for (int d = 0; d < D_; ++d) acc += sHn[d] * w1[d * DFF_ + j];
    sHid[j] = fmaxf(acc, 0.f);
  }
  __syncthreads();
  if (tid < D_) {
    float acc = b2[tid] + sHres[tid];
    for (int j = 0; j < DFF_; ++j) acc += sHid[j] * w2[j * D_ + tid];
    sDec[tid] = acc;
  }
  __syncthreads();
  if (tid < D_) {
    float acc = 0.f;
    for (int e = 0; e < D_; ++e) acc += sDec[e] * pwq[e * D_ + tid];
    sQp[tid] = acc;
  }
  __syncthreads();

  // ---------- phase 3: pointer logits; Kp via WMMA, tiles fed by TDM if available ----------
  const float scaleP = 0.08838834764831845f;   // 1/sqrt(128)
  float lmax = -3.0e38f;

#if HAS_TDM
  const unsigned int stg_lds = (unsigned int)(uintptr_t)stg;
  if (wave == 0) tdm_load_tile(mem + ((size_t)b * G_) * D_, stg_lds);   // tile 0
#endif

  for (int g0 = 0; g0 < G_; g0 += TG) {
#if HAS_TDM
    if (wave == 0) __builtin_amdgcn_s_wait_tensorcnt(0);
    __syncthreads();                         // staging valid for all waves
#pragma unroll
    for (int i = 0; i < 16; ++i)
      sA3[r * D_ + c0 + i] = (_Float16)stg[r * D_ + c0 + i];
    __syncthreads();                         // sA3 ready; staging drained
    if (wave == 0 && g0 + TG < G_)
      tdm_load_tile(mem + ((size_t)b * G_ + g0 + TG) * D_, stg_lds);   // overlap with WMMA
#else
    {
      const float* row = mem + ((size_t)b * G_ + g0 + r) * D_;
      if (g0 + TG < G_) __builtin_prefetch(row + TG * D_, 0, 0);
#pragma unroll
      for (int i = 0; i < 16; ++i) sA3[r * D_ + c0 + i] = (_Float16)row[c0 + i];
    }
    __syncthreads();
#endif

    // Kp(32x128) = A(32x128) x ptr_wk(128x128); 16 subtiles, 2 per wave
#pragma unroll
    for (int si = 0; si < 2; ++si) {
      int s4 = wave * 2 + si;
      int m0 = (s4 & 1) * 16;
      int n0 = (s4 >> 1) * 16;
      int ncol = n0 + (lane & 15);
      v8f_t acc = wmma_tile_128(&sA3[m0 * D_], &pkt[ncol * D_], lane);
      int mrow = m0 + ((lane >> 4) * 8);
#pragma unroll
      for (int i = 0; i < 8; ++i) sKp[(mrow + i) * D_ + ncol] = acc[i];
    }
    __syncthreads();

    // Up = 10*tanh(Qp . Kp / sqrt(D)); mask -> -10000
    {
      float dot = 0.f;
#pragma unroll
      for (int i = 0; i < 16; ++i) dot += sQp[c0 + i] * sKp[r * D_ + c0 + i];
#pragma unroll
      for (int m = 4; m >= 1; m >>= 1) dot += __shfl_xor(dot, m, 32);
      float up = 10.f * tanhf(dot * scaleP);
      if (mask[(size_t)b * G_ + g0 + r]) up = -10000.f;
      if ((tid & 7) == 0) sUp[g0 + r] = up;
      lmax = fmaxf(lmax, up);
    }
    __syncthreads();
  }

  // ---------- log-softmax over G ----------
  float gmax = blockReduceMax(lmax, sRed);
  float lsum = 0.f;
  for (int i = tid; i < G_; i += 256) lsum += __expf(sUp[i] - gmax);
  float tot = blockReduceSum(lsum, sRed);
  float logZ = __logf(tot);
  for (int i = tid; i < G_; i += 256) out[(size_t)b * G_ + i] = sUp[i] - gmax - logZ;
}

// ---------------- launcher ----------------
extern "C" void kernel_launch(void* const* d_in, const int* in_sizes, int n_in,
                              void* d_out, int out_size, void* d_ws, size_t ws_size,
                              hipStream_t stream) {
  (void)in_sizes; (void)n_in; (void)out_size; (void)ws_size;
  const float* tgt  = (const float*)d_in[0];
  const float* mem  = (const float*)d_in[1];
  const unsigned char* mask = (const unsigned char*)d_in[2];
  const float* ln1w = (const float*)d_in[3];
  const float* ln1b = (const float*)d_in[4];
  const float* ln2w = (const float*)d_in[5];
  const float* ln2b = (const float*)d_in[6];
  const float* wq   = (const float*)d_in[7];
  const float* wk   = (const float*)d_in[8];
  const float* wv   = (const float*)d_in[9];
  const float* wo   = (const float*)d_in[10];
  const float* w1   = (const float*)d_in[11];
  const float* b1   = (const float*)d_in[12];
  const float* w2   = (const float*)d_in[13];
  const float* b2   = (const float*)d_in[14];
  const float* pwq  = (const float*)d_in[15];
  const float* pwk  = (const float*)d_in[16];

  _Float16* wkvt = (_Float16*)d_ws;
  _Float16* pkt  = wkvt + (size_t)NKV * D_;

  prep_weights<<<128, 256, 0, stream>>>(wk, wv, pwk, wkvt, pkt);
  attn_pointer_kernel<<<B_, 256, 0, stream>>>(tgt, mem, mask, ln1w, ln1b, ln2w, ln2b,
                                              wq, wo, w1, b1, w2, b2, pwq,
                                              wkvt, pkt, (float*)d_out);
}
